// Convolution_37976100831828
// MI455X (gfx1250) — matrix operand order repaired and checked by reading, not compiled, not run on hardware
//
#include <hip/hip_runtime.h>
#include <math.h>

#define NN 30000
#define NE 600000
#define NTILES (NN / 16)   // 1875

typedef __attribute__((ext_vector_type(16))) _Float16 v16h;
typedef __attribute__((ext_vector_type(8)))  _Float16 v8h;
typedef __attribute__((ext_vector_type(8)))  float    v8f;
typedef __attribute__((ext_vector_type(4)))  float    v4f;

#define SILU_NORM  1.679f
#define INV_SQRT3  0.57735026919f

__device__ __forceinline__ v8f wmma16(v16h a, v16h b, v8f c) {
  return __builtin_amdgcn_wmma_f32_16x16x32_f16(false, a, false, b, (short)0, c, false, false);
}

// ---------------------------------------------------------------------------
// A fragment (16x32, f16) from an f32 source, unit column stride.
// K pattern per lane: koff + {0..7} and 16 + koff + {0..7} -> two aligned
// 8-float runs, loaded as 4x float4 (32B-aligned given rstride%8==0, base%8==0).
// ---------------------------------------------------------------------------
__device__ __forceinline__ v16h load_a_c1(const float* __restrict__ src, int rstride, int lane) {
  int m = lane & 15;
  int koff = (lane >> 4) << 3;
  const float* p = src + (size_t)m * rstride + koff;
  v4f q0 = *(const v4f*)(p);
  v4f q1 = *(const v4f*)(p + 4);
  v4f q2 = *(const v4f*)(p + 16);
  v4f q3 = *(const v4f*)(p + 20);
  v16h a;
#pragma unroll
  for (int j = 0; j < 4; ++j) {
    a[j]      = (_Float16)q0[j];
    a[4 + j]  = (_Float16)q1[j];
    a[8 + j]  = (_Float16)q2[j];
    a[12 + j] = (_Float16)q3[j];
  }
  return a;
}

// A fragment with arbitrary column stride (for the interleaved x1 input).
__device__ __forceinline__ v16h load_a_s(const float* __restrict__ src, int rstride, int cstride, int lane) {
  v16h a;
  int m = lane & 15;
  int koff = (lane >> 4) << 3;
#pragma unroll
  for (int e = 0; e < 16; ++e) {
    int k = ((e >> 3) << 4) + koff + (e & 7);
    a[e] = (_Float16)src[(size_t)m * rstride + (size_t)k * cstride];
  }
  return a;
}

// B fragment: weights are pre-swizzled so each (kb, nt, lane) fragment's 16
// halves are contiguous -> one 32-byte load.
__device__ __forceinline__ v16h load_b_frag(const _Float16* __restrict__ w, int NT, int kb, int nt, int lane) {
  return *(const v16h*)(w + (size_t)(((kb * NT + nt) * 32 + lane) << 4));
}

// ---------------------------------------------------------------------------
// Weight conversion: f32 -> f16, pre-swizzled to B-fragment order.
// Half-region layout (element offsets):
//   0: sc_w0   4096: sc_w1   8192: lin1_w0  12288: lin1_w1
//   16384: lin2_w00  20480: lin2_w01  24576: lin2_w10  28672: lin2_w11
//   32768: fc_w2 (16384, pre-scaled by 1/8)
//   49152: fc_w1 (2048, padded 32x64, pre-scaled by 1/sqrt(8))
// Fragment index within a matrix: ((kb*NT + nt)*32 + lane)*16 + e, with
// k = kb*32 + 16*(e>>3) + 8*(lane>>4) + (e&7), n = nt*16 + (lane&15).
// ---------------------------------------------------------------------------
struct WPtrs {
  const float *sc0, *sc1, *l10, *l11, *l200, *l201, *l210, *l211, *fc2, *fc1;
};

__global__ void prep_weights(WPtrs p, _Float16* __restrict__ dst) {
  int i = blockIdx.x * blockDim.x + threadIdx.x;
  int e = i & 15;
  int lane = (i >> 4) & 31;
  int k_in = ((e >> 3) << 4) + ((lane >> 4) << 3) + (e & 7);   // K within a 32-row block
  int n_in = lane & 15;                // N within a 16-col tile
  if (i < 32768) {                     // eight 64x64 matrices, NT = 4
    int wsel = i >> 12;
    int frag = (i >> 9) & 7;           // kb*4 + nt
    int kb = frag >> 2, nt = frag & 3;
    int k = (kb << 5) + k_in;
    int n = (nt << 4) + n_in;
    const float* src;
    switch (wsel) {
      case 0: src = p.sc0;  break;
      case 1: src = p.sc1;  break;
      case 2: src = p.l10;  break;
      case 3: src = p.l11;  break;
      case 4: src = p.l200; break;
      case 5: src = p.l201; break;
      case 6: src = p.l210; break;
      default: src = p.l211; break;
    }
    dst[i] = (_Float16)src[k * 64 + n];
  } else if (i < 49152) {              // fc_w2: 64x256, NT = 16
    int local = i - 32768;
    int frag = local >> 9;             // kb*16 + nt
    int kb = frag >> 4, nt = frag & 15;
    int k = (kb << 5) + k_in;
    int n = (nt << 4) + n_in;
    dst[i] = (_Float16)(p.fc2[k * 256 + n] * 0.125f);
  } else if (i < 51200) {              // fc_w1: 8x64 padded to 32x64, NT = 4, kb = 0
    int local = i - 49152;
    int nt = local >> 9;               // 0..3
    int n = (nt << 4) + n_in;
    dst[i] = (k_in < 8) ? (_Float16)(p.fc1[k_in * 64 + n] * 0.35355339059f)
                        : (_Float16)0.f;
  }
}

// ---------------------------------------------------------------------------
// Node lin1: y0 = x0 @ lin1_w0 * a/8 ; y1[:,:,c] = x1[:,:,c] @ lin1_w1 * a/8
// y layout matches node_input: [y0(64) | y1 interleaved u*3+c (192)]
// 4 waves per block, one 16-node tile per wave.
// ---------------------------------------------------------------------------
__global__ void node_lin1(const float* __restrict__ xin, const float* __restrict__ attr,
                          const _Float16* __restrict__ wh, float* __restrict__ y) {
  int tile = blockIdx.x * 4 + (threadIdx.x >> 5);
  if (tile >= NTILES) return;
  int base = tile * 16;
  int lane = threadIdx.x & 31;
  int hi = lane >> 4, col = lane & 15;
  const _Float16* w0 = wh + 8192;
  const _Float16* w1 = wh + 12288;
  const float* xrow = xin + (size_t)base * 256;

  float scale[8];
#pragma unroll
  for (int r = 0; r < 8; ++r) scale[r] = attr[base + r + hi * 8] * 0.125f;

  v16h a0 = load_a_c1(xrow,      256, lane);
  v16h a1 = load_a_c1(xrow + 32, 256, lane);
#pragma unroll
  for (int nt = 0; nt < 4; ++nt) {
    v8f acc = {};
    acc = wmma16(a0, load_b_frag(w0, 4, 0, nt, lane), acc);
    acc = wmma16(a1, load_b_frag(w0, 4, 1, nt, lane), acc);
#pragma unroll
    for (int r = 0; r < 8; ++r)
      y[(size_t)(base + r + hi * 8) * 256 + nt * 16 + col] = acc[r] * scale[r];
  }
#pragma unroll
  for (int c = 0; c < 3; ++c) {
    v16h b0 = load_a_s(xrow + 64 + c,      256, 3, lane);
    v16h b1 = load_a_s(xrow + 64 + 96 + c, 256, 3, lane);
#pragma unroll
    for (int nt = 0; nt < 4; ++nt) {
      v8f acc = {};
      acc = wmma16(b0, load_b_frag(w1, 4, 0, nt, lane), acc);
      acc = wmma16(b1, load_b_frag(w1, 4, 1, nt, lane), acc);
#pragma unroll
      for (int r = 0; r < 8; ++r)
        y[(size_t)(base + r + hi * 8) * 256 + 64 + (nt * 16 + col) * 3 + c] = acc[r] * scale[r];
    }
  }
}

// ---------------------------------------------------------------------------
// Fused edge kernel, all-WMMA MLP:
//   GEMM1 (K padded 8->32): pre = es @ (fc_w1/sqrt8)        [4 WMMA]
//   silu(pre)*1.679 -> f16 h-tile in LDS -> A fragments      [ds bounce]
//   GEMM2: w = h @ (fc_w2/8)                                 [32 WMMA]
//   gather y[src] (vectorized b128) -> mids -> atomic scatter into agg[dst]
// 2 waves/block, 16 edges/wave.
// agg row layout (planar): [mid0(64) | mid1(64) | mid2 c*64+u (192) | mid3 c*64+u (192)]
// ---------------------------------------------------------------------------
__global__ void edge_kernel(const int* __restrict__ esrc, const int* __restrict__ edst,
                            const float* __restrict__ eattr, const float* __restrict__ escal,
                            const _Float16* __restrict__ wh,
                            const float* __restrict__ y, float* __restrict__ agg) {
  __shared__ float    wt[2][16][256];
  __shared__ _Float16 ht[2][16][64];
  int wave = threadIdx.x >> 5;
  int lane = threadIdx.x & 31;
  int ebase = (blockIdx.x * 2 + wave) * 16;
  int m = lane & 15, hi = lane >> 4;
  const _Float16* fc2h = wh + 32768;
  const _Float16* fc1h = wh + 49152;

  // A fragment for es (16x32, only K<8 nonzero -> lanes 0..15, elements 0..7)
  v16h Aes;
#pragma unroll
  for (int e2 = 0; e2 < 16; ++e2) Aes[e2] = (_Float16)0.f;
  {
    const float* er = escal + (size_t)(ebase + m) * 8;
    v4f q0 = *(const v4f*)(er);
    v4f q1 = *(const v4f*)(er + 4);
    if (hi == 0) {
#pragma unroll
      for (int j = 0; j < 4; ++j) {
        Aes[j]     = (_Float16)q0[j];
        Aes[4 + j] = (_Float16)q1[j];
      }
    }
  }

  // GEMM1 + silu -> h tile (f16) in LDS
#pragma unroll
  for (int nt = 0; nt < 4; ++nt) {
    v8f zero = {};
    v8f pre = wmma16(Aes, load_b_frag(fc1h, 4, 0, nt, lane), zero);
#pragma unroll
    for (int r = 0; r < 8; ++r) {
      float v = pre[r];
      float s = v / (1.f + __expf(-v)) * SILU_NORM;   // silu * norm
      ht[wave][r + hi * 8][nt * 16 + m] = (_Float16)s;
    }
  }
  __syncthreads();

  // Rebuild GEMM2 A fragments from LDS (4x aligned 16B ds loads per fragment pair)
  v16h A[2];
  {
    const _Float16* hrow = &ht[wave][m][0];
    int koff = hi << 3;
#pragma unroll
    for (int kb = 0; kb < 2; ++kb) {
      v8h lo = *(const v8h*)(hrow + kb * 32 + koff);
      v8h up = *(const v8h*)(hrow + kb * 32 + 16 + koff);
#pragma unroll
      for (int j = 0; j < 8; ++j) { A[kb][j] = lo[j]; A[kb][8 + j] = up[j]; }
    }
  }

  // GEMM2: w = h @ (fc_w2/8) : 16 N-tiles x 2 K-blocks of WMMA
#pragma unroll
  for (int nt = 0; nt < 16; ++nt) {
    v8f acc = {};
    acc = wmma16(A[0], load_b_frag(fc2h, 16, 0, nt, lane), acc);
    acc = wmma16(A[1], load_b_frag(fc2h, 16, 1, nt, lane), acc);
#pragma unroll
    for (int r = 0; r < 8; ++r) wt[wave][r + hi * 8][nt * 16 + m] = acc[r];
  }
  __syncthreads();

  // Scatter: each lane handles half (32 u-values) of one edge, in chunks of 4
  // so all y / LDS traffic is b128.
  int e = ebase + m;
  int src = esrc[e], dst = edst[e];
  v4f ea = *(const v4f*)(eattr + (size_t)e * 4);
  float ea0 = ea[0], e1x = ea[1], e1y = ea[2], e1z = ea[3];
  int u0 = hi * 32;
  const float* yq = y + (size_t)src * 256 + u0;       // x0 gather base
  const float* yg = y + (size_t)src * 256 + 64 + 3 * u0;  // vector gather base
  float* ar = agg + (size_t)dst * 512 + u0;
  const float* wq = wt[wave][m] + u0;
#pragma unroll 2
  for (int cu = 0; cu < 8; ++cu) {
    int ub = cu * 4;
    v4f x0v = *(const v4f*)(yq + ub);
    v4f ga  = *(const v4f*)(yg + ub * 3);
    v4f gb  = *(const v4f*)(yg + ub * 3 + 4);
    v4f gc  = *(const v4f*)(yg + ub * 3 + 8);
    v4f wav = *(const v4f*)(wq + ub);
    v4f wbv = *(const v4f*)(wq + 64 + ub);
    v4f wcv = *(const v4f*)(wq + 128 + ub);
    v4f wdv = *(const v4f*)(wq + 192 + ub);
    float g[12];
#pragma unroll
    for (int j = 0; j < 4; ++j) { g[j] = ga[j]; g[4 + j] = gb[j]; g[8 + j] = gc[j]; }
#pragma unroll
    for (int j = 0; j < 4; ++j) {
      int u = ub + j;
      float x0u = x0v[j];
      float g0 = g[3 * j], g1 = g[3 * j + 1], g2 = g[3 * j + 2];
      atomicAdd(ar + u,       wav[j] * x0u * ea0);
      atomicAdd(ar + 64 + u,  wbv[j] * (g0 * e1x + g1 * e1y + g2 * e1z) * INV_SQRT3);
      float t = wcv[j] * x0u;
      atomicAdd(ar + 128 + u, t * e1x);
      atomicAdd(ar + 192 + u, t * e1y);
      atomicAdd(ar + 256 + u, t * e1z);
      float t2 = wdv[j] * ea0;
      atomicAdd(ar + 320 + u, t2 * g0);
      atomicAdd(ar + 384 + u, t2 * g1);
      atomicAdd(ar + 448 + u, t2 * g2);
    }
  }
}

// ---------------------------------------------------------------------------
// Final: self-connection + lin2 on agg, combined with sin/cos(pi/8).
// 4 waves per block, one 16-node tile per wave. agg planar -> unit-stride loads.
// ---------------------------------------------------------------------------
__global__ void node_final(const float* __restrict__ xin, const float* __restrict__ attr,
                           const float* __restrict__ agg, const _Float16* __restrict__ wh,
                           float* __restrict__ out) {
  int tile = blockIdx.x * 4 + (threadIdx.x >> 5);
  if (tile >= NTILES) return;
  int base = tile * 16;
  int lane = threadIdx.x & 31;
  int hi = lane >> 4, col = lane & 15;
  const _Float16* sc0 = wh;
  const _Float16* sc1 = wh + 4096;
  const _Float16* w00 = wh + 16384;
  const _Float16* w01 = wh + 20480;
  const _Float16* w10 = wh + 24576;
  const _Float16* w11 = wh + 28672;
  const float* xrow = xin + (size_t)base * 256;
  const float* grow = agg + (size_t)base * 512;
  const float c_s = 0.38268343236f;  // sin(pi/8)
  const float c_x = 0.92387953251f;  // cos(pi/8)
  const float inv2nn = 0.08838834764f * 0.22360679775f;  // 1/sqrt(128) * 1/sqrt(20)

  float sa[8], so[8];
#pragma unroll
  for (int r = 0; r < 8; ++r) {
    float a = attr[base + r + hi * 8];
    sa[r] = c_s * a * 0.125f;
    so[r] = c_x * a * inv2nn;
  }

  // scalar channel
  v16h ax0 = load_a_c1(xrow,      256, lane);
  v16h ax1 = load_a_c1(xrow + 32, 256, lane);
  v16h za0 = load_a_c1(grow,      512, lane);
  v16h za1 = load_a_c1(grow + 32, 512, lane);
  v16h zb0 = load_a_c1(grow + 64, 512, lane);
  v16h zb1 = load_a_c1(grow + 96, 512, lane);
#pragma unroll
  for (int nt = 0; nt < 4; ++nt) {
    v8f s = {};
    s = wmma16(ax0, load_b_frag(sc0, 4, 0, nt, lane), s);
    s = wmma16(ax1, load_b_frag(sc0, 4, 1, nt, lane), s);
    v8f o = {};
    o = wmma16(za0, load_b_frag(w00, 4, 0, nt, lane), o);
    o = wmma16(za1, load_b_frag(w00, 4, 1, nt, lane), o);
    o = wmma16(zb0, load_b_frag(w01, 4, 0, nt, lane), o);
    o = wmma16(zb1, load_b_frag(w01, 4, 1, nt, lane), o);
#pragma unroll
    for (int r = 0; r < 8; ++r)
      out[(size_t)(base + r + hi * 8) * 256 + nt * 16 + col] = sa[r] * s[r] + so[r] * o[r];
  }

  // vector channels (agg planar -> unit-stride loads; x1 stays interleaved)
#pragma unroll
  for (int c = 0; c < 3; ++c) {
    v16h x1a  = load_a_s(xrow + 64 + c,      256, 3, lane);
    v16h x1b  = load_a_s(xrow + 64 + 96 + c, 256, 3, lane);
    v16h z1a0 = load_a_c1(grow + 128 + c * 64,      512, lane);
    v16h z1a1 = load_a_c1(grow + 128 + c * 64 + 32, 512, lane);
    v16h z1b0 = load_a_c1(grow + 320 + c * 64,      512, lane);
    v16h z1b1 = load_a_c1(grow + 320 + c * 64 + 32, 512, lane);
#pragma unroll
    for (int nt = 0; nt < 4; ++nt) {
      v8f s = {};
      s = wmma16(x1a, load_b_frag(sc1, 4, 0, nt, lane), s);
      s = wmma16(x1b, load_b_frag(sc1, 4, 1, nt, lane), s);
      v8f o = {};
      o = wmma16(z1a0, load_b_frag(w10, 4, 0, nt, lane), o);
      o = wmma16(z1a1, load_b_frag(w10, 4, 1, nt, lane), o);
      o = wmma16(z1b0, load_b_frag(w11, 4, 0, nt, lane), o);
      o = wmma16(z1b1, load_b_frag(w11, 4, 1, nt, lane), o);
#pragma unroll
      for (int r = 0; r < 8; ++r)
        out[(size_t)(base + r + hi * 8) * 256 + 64 + (nt * 16 + col) * 3 + c] = sa[r] * s[r] + so[r] * o[r];
    }
  }
}

extern "C" void kernel_launch(void* const* d_in, const int* in_sizes, int n_in,
                              void* d_out, int out_size, void* d_ws, size_t ws_size,
                              hipStream_t stream) {
  const float* node_input = (const float*)d_in[0];
  const float* node_attr  = (const float*)d_in[1];
  const int*   edge_src   = (const int*)d_in[2];
  const int*   edge_dst   = (const int*)d_in[3];
  const float* edge_attr  = (const float*)d_in[4];
  const float* edge_scal  = (const float*)d_in[5];
  const float* sc_w0      = (const float*)d_in[6];
  const float* sc_w1      = (const float*)d_in[7];
  const float* lin1_w0    = (const float*)d_in[8];
  const float* lin1_w1    = (const float*)d_in[9];
  const float* fc_w1      = (const float*)d_in[10];
  const float* fc_w2      = (const float*)d_in[11];
  const float* l200       = (const float*)d_in[12];
  const float* l201       = (const float*)d_in[13];
  const float* l210       = (const float*)d_in[14];
  const float* l211       = (const float*)d_in[15];

  float*    agg = (float*)d_ws;                        // N*512 f32 (61.44 MB)
  float*    y   = agg + (size_t)NN * 512;              // N*256 f32 (30.72 MB)
  _Float16* wh  = (_Float16*)(y + (size_t)NN * 256);   // 51200 halfs, swizzled

  hipMemsetAsync(agg, 0, (size_t)NN * 512 * sizeof(float), stream);

  WPtrs p{sc_w0, sc_w1, lin1_w0, lin1_w1, l200, l201, l210, l211, fc_w2, fc_w1};
  prep_weights<<<200, 256, 0, stream>>>(p, wh);
  node_lin1<<<(NTILES + 3) / 4, 128, 0, stream>>>(node_input, node_attr, wh, y);
  edge_kernel<<<NE / 32, 64, 0, stream>>>(edge_src, edge_dst, edge_attr, edge_scal,
                                          wh, y, agg);
  node_final<<<(NTILES + 3) / 4, 128, 0, stream>>>(node_input, node_attr, agg, wh, (float*)d_out);
}
